// GatingNetwork_19885698581126
// MI455X (gfx1250) — matrix operand order repaired and checked by reading, not compiled, not run on hardware
//
#include <hip/hip_runtime.h>
#include <hip/hip_bf16.h>

typedef __attribute__((ext_vector_type(16))) _Float16 v16h;
typedef __attribute__((ext_vector_type(8)))  _Float16 v8h;
typedef __attribute__((ext_vector_type(8)))  float    v8f;

#define T_LEN 200
#define NBINS 40

// ---------------------------------------------------------------------------
// Kernel 1: per-sample histogram features -> padded fp16 rows [B][64] in d_ws
// One wave32 per sample. Lanes load int4 (global_load_b128, rows are 16B
// aligned: 200*4 = 800 bytes), bin via LDS atomics (ds_add_u32), reduce
// seq_len/unique via shfl_xor. This phase is the HBM roofline bottleneck
// (52.4 MB of tokens ~= 2.6 us at 23.3 TB/s), so wide coalesced loads matter.
// ---------------------------------------------------------------------------
__launch_bounds__(256)
__global__ void feat_kernel(const int* __restrict__ x, _Float16* __restrict__ feat, int B) {
    __shared__ unsigned hist[8][NBINS];
    const int wave = threadIdx.x >> 5;
    const int lane = threadIdx.x & 31;
    const int s = blockIdx.x * 8 + wave;
    if (s >= B) return;  // wave-uniform exit

    // zero the wave's 40 bins (lanes 0..31 -> bins 0..31, lanes 0..7 -> 32..39)
    hist[wave][lane] = 0u;
    if (lane < NBINS - 32) hist[wave][32 + lane] = 0u;
    __builtin_amdgcn_wave_barrier();

    // 200 tokens = 50 int4 per row: pass 1 = int4[lane], pass 2 = int4[32+lane]
    const int4* row4 = (const int4*)(x + (long)s * T_LEN);
    {
        int4 va = row4[lane];
        if (va.x > 0) atomicAdd(&hist[wave][va.x], 1u);   // ds_add_u32
        if (va.y > 0) atomicAdd(&hist[wave][va.y], 1u);
        if (va.z > 0) atomicAdd(&hist[wave][va.z], 1u);
        if (va.w > 0) atomicAdd(&hist[wave][va.w], 1u);
    }
    if (lane < (T_LEN / 4 - 32)) {   // 18 lanes cover int4 32..49
        int4 vb = row4[32 + lane];
        if (vb.x > 0) atomicAdd(&hist[wave][vb.x], 1u);
        if (vb.y > 0) atomicAdd(&hist[wave][vb.y], 1u);
        if (vb.z > 0) atomicAdd(&hist[wave][vb.z], 1u);
        if (vb.w > 0) atomicAdd(&hist[wave][vb.w], 1u);
    }
    __builtin_amdgcn_wave_barrier();

    unsigned ca = hist[wave][lane];
    unsigned cb = (lane < 8) ? hist[wave][32 + lane] : 0u;
    unsigned scnt = ca + cb;                         // nonzero-token count
    unsigned ucnt = (ca > 0u ? 1u : 0u) + (cb > 0u ? 1u : 0u);
    for (int off = 16; off > 0; off >>= 1) {
        scnt += __shfl_xor(scnt, off, 32);
        ucnt += __shfl_xor(ucnt, off, 32);
    }

    _Float16* frow = feat + (long)s * 64;
    // feature layout: [seq_len, unique, hist[0..39], 0-pad to 64]
    float f0;
    if (lane == 0)      f0 = (float)scnt;
    else if (lane == 1) f0 = (float)ucnt;
    else                f0 = (float)hist[wave][lane - 2];
    frow[lane] = (_Float16)f0;
    int c = lane + 32;
    float f1 = (c < 42) ? (float)hist[wave][c - 2] : 0.0f;
    frow[c] = (_Float16)f1;
}

// ---------------------------------------------------------------------------
// WMMA fragment helpers (wave32, 16x16x32 f16; A/B use this file's contiguous k order)
// A (16x32): lane holds row m = lane&15; f16 slot j (0..15):
//   K = kb*32 + (laneHi ? 16 : 0) + j  -> one contiguous run of 16
// B (32x16): lane holds col N = lane&15; slot j: K = kb*32 + (laneHi?16:0) + j
// C/D (16x16 f32): vgpr r: M = r + (laneHi?8:0), N = lane&15
// ---------------------------------------------------------------------------
__device__ __forceinline__ v16h combine8(v8h lo, v8h hi8) {
    v16h a;
#pragma unroll
    for (int i = 0; i < 8; ++i) { a[i] = lo[i]; a[i + 8] = hi8[i]; }
    return a;
}

__device__ __forceinline__ v16h load_afrag(const _Float16* base, int ld, int m, int k0, int hi) {
    const v8h* p = (const v8h*)(base + (long)m * ld + k0 + hi * 16);
    return combine8(p[0], p[1]);
}

__device__ __forceinline__ v16h load_bfrag(const _Float16* w, int f, int lane) {
    const v8h* p = (const v8h*)(w + ((f * 32 + lane) << 4));
    return combine8(p[0], p[1]);
}

// ---------------------------------------------------------------------------
// Kernel 2: fused MLP, 4 waves/block, one 16-sample tile per wave iteration.
// Weights staged to LDS once per block (pre-swizzled into B-fragment order),
// then hoisted into registers: all 34 B-fragments (272 VGPRs, using the CDNA5
// extended VGPR file) stay resident across the tile loop, so the loop body is
// A-loads + 34 back-to-back WMMAs + the inter-layer LDS re-layout only.
// ---------------------------------------------------------------------------
__launch_bounds__(128)
__global__ void mlp_kernel(const _Float16* __restrict__ feat,
                           const float* __restrict__ W1, const float* __restrict__ b1,
                           const float* __restrict__ W2, const float* __restrict__ b2,
                           const float* __restrict__ W3, const float* __restrict__ b3,
                           float* __restrict__ out, int numTiles) {
    __shared__ __align__(16) _Float16 w1f[8 * 2 * 32 * 16];  // 8 n-tiles x 2 kb
    __shared__ __align__(16) _Float16 w2f[4 * 4 * 32 * 16];  // 4 n-tiles x 4 kb
    __shared__ __align__(16) _Float16 w3f[2 * 32 * 16];      // 1 n-tile  x 2 kb
    __shared__ __align__(16) _Float16 h1buf[4][16 * 128];    // per-wave
    __shared__ __align__(16) _Float16 h2buf[4][16 * 64];     // per-wave

    const int tid = threadIdx.x;

    // ---- stage weights into WMMA B-fragment order (f32 -> f16) ----
    for (int idx = tid; idx < 8 * 2 * 32 * 16; idx += 128) {
        int j = idx & 15, ln = (idx >> 4) & 31, f = idx >> 9;
        int n = f >> 1, kb = f & 1, N = ln & 15, hi = ln >> 4;
        int k = kb * 32 + hi * 16 + j;
        w1f[idx] = (_Float16)((k < 42) ? W1[k * 128 + n * 16 + N] : 0.0f);
    }
    for (int idx = tid; idx < 4 * 4 * 32 * 16; idx += 128) {
        int j = idx & 15, ln = (idx >> 4) & 31, f = idx >> 9;
        int n = f >> 2, kb = f & 3, N = ln & 15, hi = ln >> 4;
        int k = kb * 32 + hi * 16 + j;
        w2f[idx] = (_Float16)W2[k * 64 + n * 16 + N];
    }
    for (int idx = tid; idx < 2 * 32 * 16; idx += 128) {
        int j = idx & 15, ln = (idx >> 4) & 31, kb = idx >> 9;
        int N = ln & 15, hi = ln >> 4;
        int k = kb * 32 + hi * 16 + j;
        w3f[idx] = (_Float16)((N < 3) ? W3[k * 3 + N] : 0.0f);
    }
    __syncthreads();

    const int wave = tid >> 5, lane = tid & 31;
    const int N = lane & 15, hi = lane >> 4;
    const int m = N;  // A-fragment row index for this lane
    _Float16* h1 = h1buf[wave];
    _Float16* h2 = h2buf[wave];
    const int wavesTotal = gridDim.x * 4;

    // ---- hoist all B fragments + biases into registers (loop-invariant) ----
    v16h bw1[16], bw2[16], bw3[2];
#pragma unroll
    for (int f = 0; f < 16; ++f) bw1[f] = load_bfrag(w1f, f, lane);
#pragma unroll
    for (int f = 0; f < 16; ++f) bw2[f] = load_bfrag(w2f, f, lane);
    bw3[0] = load_bfrag(w3f, 0, lane);
    bw3[1] = load_bfrag(w3f, 1, lane);

    float bias1[8], bias2[4];
#pragma unroll
    for (int n = 0; n < 8; ++n) bias1[n] = b1[n * 16 + N];
#pragma unroll
    for (int n = 0; n < 4; ++n) bias2[n] = b2[n * 16 + N];
    const float bias3 = (N < 3) ? b3[N] : 0.0f;

    for (int tile = blockIdx.x * 4 + wave; tile < numTiles; tile += wavesTotal) {
        const _Float16* frows = feat + (long)tile * 16 * 64;

        // ---- layer 1: [16x64] x [64x128] ----
        v16h a0 = load_afrag(frows, 64, m, 0, hi);
        v16h a1 = load_afrag(frows, 64, m, 32, hi);
#pragma unroll
        for (int n = 0; n < 8; ++n) {
            v8f c = {};
            c = __builtin_amdgcn_wmma_f32_16x16x32_f16(false, a0, false,
                    bw1[n * 2 + 0], (short)0, c, false, false);
            c = __builtin_amdgcn_wmma_f32_16x16x32_f16(false, a1, false,
                    bw1[n * 2 + 1], (short)0, c, false, false);
#pragma unroll
            for (int r = 0; r < 8; ++r) {
                float h = c[r] + bias1[n];
                h = h > 0.0f ? h : 0.0f;
                h1[(r + hi * 8) * 128 + n * 16 + N] = (_Float16)h;
            }
        }
        __builtin_amdgcn_wave_barrier();  // same-wave LDS ops are in-order

        // ---- layer 2: [16x128] x [128x64] ----
        v16h a2[4];
#pragma unroll
        for (int kb = 0; kb < 4; ++kb) a2[kb] = load_afrag(h1, 128, m, kb * 32, hi);
#pragma unroll
        for (int n = 0; n < 4; ++n) {
            v8f c = {};
#pragma unroll
            for (int kb = 0; kb < 4; ++kb)
                c = __builtin_amdgcn_wmma_f32_16x16x32_f16(false, a2[kb], false,
                        bw2[n * 4 + kb], (short)0, c, false, false);
#pragma unroll
            for (int r = 0; r < 8; ++r) {
                float h = c[r] + bias2[n];
                h = h > 0.0f ? h : 0.0f;
                h2[(r + hi * 8) * 64 + n * 16 + N] = (_Float16)h;
            }
        }
        __builtin_amdgcn_wave_barrier();

        // ---- layer 3: [16x64] x [64x16(pad of 3)] + softmax over 3 ----
        v16h a30 = load_afrag(h2, 64, m, 0, hi);
        v16h a31 = load_afrag(h2, 64, m, 32, hi);
        v8f c = {};
        c = __builtin_amdgcn_wmma_f32_16x16x32_f16(false, a30, false,
                bw3[0], (short)0, c, false, false);
        c = __builtin_amdgcn_wmma_f32_16x16x32_f16(false, a31, false,
                bw3[1], (short)0, c, false, false);
#pragma unroll
        for (int r = 0; r < 8; ++r) {
            float val = c[r] + bias3;
            int base = hi << 4;
            float v0 = __shfl(val, base + 0, 32);
            float v1 = __shfl(val, base + 1, 32);
            float v2 = __shfl(val, base + 2, 32);
            float mx = fmaxf(v0, fmaxf(v1, v2));
            float e0 = __expf(v0 - mx), e1 = __expf(v1 - mx), e2 = __expf(v2 - mx);
            float inv = 1.0f / (e0 + e1 + e2);
            if (N < 3) {
                int rowIdx = tile * 16 + r + hi * 8;
                out[rowIdx * 3 + N] = (N == 0 ? e0 : (N == 1 ? e1 : e2)) * inv;
            }
        }
    }
}

// ---------------------------------------------------------------------------
// Launcher. Inputs (setup_inputs order): x, W1, b1, W2, b2, W3, b3.
// d_ws holds the padded fp16 feature matrix [B][64] (8 MB for B=65536).
// ---------------------------------------------------------------------------
extern "C" void kernel_launch(void* const* d_in, const int* in_sizes, int n_in,
                              void* d_out, int out_size, void* d_ws, size_t ws_size,
                              hipStream_t stream) {
    const int*   x  = (const int*)d_in[0];
    const float* W1 = (const float*)d_in[1];
    const float* b1 = (const float*)d_in[2];
    const float* W2 = (const float*)d_in[3];
    const float* b2 = (const float*)d_in[4];
    const float* W3 = (const float*)d_in[5];
    const float* b3 = (const float*)d_in[6];
    float* out = (float*)d_out;

    int B = in_sizes[0] / T_LEN;
    _Float16* feat = (_Float16*)d_ws;

    int blocks1 = (B + 7) / 8;  // 8 waves (samples) per 256-thread block
    feat_kernel<<<blocks1, 256, 0, stream>>>(x, feat, B);

    int numTiles = B / 16;
    mlp_kernel<<<128, 128, 0, stream>>>(feat, W1, b1, W2, b2, W3, b3, out, numTiles);
}
